// ResidualBlock_26645977104500
// MI455X (gfx1250) — matrix operand order repaired and checked by reading, not compiled, not run on hardware
//
#include <hip/hip_runtime.h>
#include <math.h>

#define D_MODEL 1024
#define D_INNER 2048
#define D_STATE 16
#define DT_RANK 64
#define D_CONV  4
#define SEQ     2048
#define EPSV    1e-5f
#define XPROJ_N (DT_RANK + 2 * D_STATE)   // 96

typedef __attribute__((ext_vector_type(16))) __bf16 v16bf;
typedef __attribute__((ext_vector_type(8)))  __bf16 v8bf;
typedef __attribute__((ext_vector_type(8)))  float  v8f;

// ---------------------------------------------------------------------------
// Transpose + convert: src[K,N] f32  ->  dst[N,K] bf16  (N-major "Bt" layout)
// LDS-tiled 32x32 so both global read and write are coalesced.
// ---------------------------------------------------------------------------
__global__ void transpose_to_bf16_kernel(const float* __restrict__ src,
                                         __bf16* __restrict__ dst,
                                         int K, int N) {
  __shared__ __bf16 tile[32][33];
  const int kb = blockIdx.y * 32, nb = blockIdx.x * 32;
  const int tx = threadIdx.x & 31, ty = threadIdx.x >> 5;   // 32 x 8
  #pragma unroll
  for (int i = ty; i < 32; i += 8)
    tile[i][tx] = (__bf16)src[(size_t)(kb + i) * N + nb + tx];
  __syncthreads();
  #pragma unroll
  for (int i = ty; i < 32; i += 8)
    dst[(size_t)(nb + i) * K + kb + tx] = tile[tx][i];
}

// strided slice convert: dst[r*cols + c] = src[r*src_ld + c]  (dlt -> bf16)
__global__ void slice_f32_to_bf16_kernel(const float* __restrict__ src,
                                         __bf16* __restrict__ dst,
                                         int rows, int src_ld, int cols) {
  int i = blockIdx.x * blockDim.x + threadIdx.x;
  if (i >= rows * cols) return;
  int r = i / cols, c = i % cols;
  dst[i] = (__bf16)src[(size_t)r * src_ld + c];
}

// ---------------------------------------------------------------------------
// RMSNorm: one block per sequence position; h = x * rsqrt(mean(x^2)+eps) * w
// ---------------------------------------------------------------------------
__global__ void rmsnorm_kernel(const float* __restrict__ x,
                               const float* __restrict__ w,
                               __bf16* __restrict__ h) {
  const int row = blockIdx.x;
  const float* xr = x + (size_t)row * D_MODEL;
  float ss = 0.f;
  for (int i = threadIdx.x; i < D_MODEL; i += blockDim.x) {
    float v = xr[i];
    ss += v * v;
  }
  #pragma unroll
  for (int off = 16; off > 0; off >>= 1) ss += __shfl_xor(ss, off, 32);
  __shared__ float red[8];
  int wave = threadIdx.x >> 5, lane = threadIdx.x & 31;
  if (lane == 0) red[wave] = ss;
  __syncthreads();
  if (threadIdx.x == 0) {
    float t = 0.f;
    #pragma unroll
    for (int i = 0; i < 8; ++i) t += red[i];
    red[0] = rsqrtf(t / (float)D_MODEL + EPSV);
  }
  __syncthreads();
  float scale = red[0];
  for (int i = threadIdx.x; i < D_MODEL; i += blockDim.x)
    h[(size_t)row * D_MODEL + i] = (__bf16)(xr[i] * scale * w[i]);
}

// ---------------------------------------------------------------------------
// WMMA bf16 GEMM: C[M,N] = A[M,K] (row-major bf16) x Bt[N,K] (N-major bf16).
// Per CDNA5 ISA 7.12.2 the per-lane A fragment is two contiguous 16B runs of
// the row-major A (K = hi*8..+7 and 16+hi*8..+7) and the per-lane B fragment
// is the same two 16B runs of Bt (same K as A): all loads are b128.
// Each wave computes a 16x32 tile (A fragment reused across two WMMAs).
// EPI: 0 = store, 1 = softplus(acc + bias[n]), 2 = acc + resid[m*N+n].
// ---------------------------------------------------------------------------
template <int EPI>
__global__ void wmma_gemm_bf16_kernel(const __bf16* __restrict__ A,
                                      const __bf16* __restrict__ Bt,
                                      float* __restrict__ C,
                                      int M, int N, int K,
                                      const float* __restrict__ bias,
                                      const float* __restrict__ resid) {
  const int lane  = threadIdx.x & 31;
  const int wave  = threadIdx.x >> 5;
  const int l     = lane & 15;
  const int hi    = (lane >> 4) & 1;
  const int nTile = blockIdx.x * 128 + wave * 32;   // 4 waves * 32 cols
  const int mTile = blockIdx.y * 16;
  if (nTile >= N) return;
  const bool has1 = (nTile + 16) < N;

  const __bf16* Ap  = A  + (size_t)(mTile + l) * K;
  const __bf16* Bp0 = Bt + (size_t)(nTile + l) * K;
  const __bf16* Bp1 = Bt + (size_t)(has1 ? (nTile + 16 + l) : (nTile + l)) * K;

  v8f acc0 = {}, acc1 = {};
  for (int k0 = 0; k0 < K; k0 += 32) {
    __builtin_prefetch(Ap + k0 + 128, 0, 3);          // global_prefetch_b8
    __builtin_prefetch(Bp0 + k0 + 128, 0, 3);
    v8bf a_lo = *(const v8bf*)(Ap + k0 + hi * 8);         // b128
    v8bf a_hi = *(const v8bf*)(Ap + k0 + 16 + hi * 8);    // b128
    v16bf a = __builtin_shufflevector(a_lo, a_hi,
        0, 1, 2, 3, 4, 5, 6, 7, 8, 9, 10, 11, 12, 13, 14, 15);
    v16bf b0 = __builtin_shufflevector(*(const v8bf*)(Bp0 + k0 + hi * 8), *(const v8bf*)(Bp0 + k0 + 16 + hi * 8),
        0, 1, 2, 3, 4, 5, 6, 7, 8, 9, 10, 11, 12, 13, 14, 15);       // 2 x b128
    v16bf b1 = __builtin_shufflevector(*(const v8bf*)(Bp1 + k0 + hi * 8), *(const v8bf*)(Bp1 + k0 + 16 + hi * 8),
        0, 1, 2, 3, 4, 5, 6, 7, 8, 9, 10, 11, 12, 13, 14, 15);       // 2 x b128
    acc0 = __builtin_amdgcn_wmma_f32_16x16x32_bf16(
        false, a, false, b0, (short)0, acc0, false, false);
    acc1 = __builtin_amdgcn_wmma_f32_16x16x32_bf16(
        false, a, false, b1, (short)0, acc1, false, false);
  }

  #pragma unroll
  for (int r = 0; r < 8; ++r) {
    const int mo = mTile + r + hi * 8;
    #pragma unroll
    for (int q = 0; q < 2; ++q) {
      if (q == 1 && !has1) break;
      const int no = nTile + q * 16 + l;
      float v = (q == 0) ? acc0[r] : acc1[r];
      if (EPI == 1) {                 // stable softplus(v + bias), branchless
        v += bias[no];
        v = fmaxf(v, 0.f) + log1pf(__expf(-fabsf(v)));
      } else if (EPI == 2) {          // + residual
        v += resid[(size_t)mo * N + no];
      }
      C[(size_t)mo * N + no] = v;
    }
  }
}

// ---------------------------------------------------------------------------
// Depthwise causal conv (width 4, left pad 3) + bias + SiLU.
// Reads the xc half of xr (row stride 2*D_INNER); writes f32 + bf16 copies.
// ---------------------------------------------------------------------------
__global__ void conv_silu_kernel(const float* __restrict__ xr,
                                 const float* __restrict__ convW,
                                 const float* __restrict__ convb,
                                 float* __restrict__ xc,
                                 __bf16* __restrict__ xcbf) {
  int idx = blockIdx.x * blockDim.x + threadIdx.x;
  if (idx >= SEQ * D_INNER) return;
  int d = idx % D_INNER;
  int t = idx / D_INNER;
  float acc = convb[d];
  #pragma unroll
  for (int wtap = 0; wtap < D_CONV; ++wtap) {
    int tt = t - (D_CONV - 1) + wtap;
    if (tt >= 0)
      acc += convW[wtap * D_INNER + d] * xr[(size_t)tt * (2 * D_INNER) + d];
  }
  float sv = acc / (1.f + __expf(-acc));   // silu
  xc[idx]   = sv;
  xcbf[idx] = (__bf16)sv;
}

// ---------------------------------------------------------------------------
// Selective scan. Thread = (channel d, state n); 16-lane group per channel.
// Per-channel scalars (delta, u, res) are loaded by lane n==0 only and
// broadcast with shfl(width=16); state lives in registers for all 2048 steps.
// ---------------------------------------------------------------------------
__global__ void scan_kernel(const float* __restrict__ delta,
                            const float* __restrict__ xc,
                            const float* __restrict__ xdbl,
                            const float* __restrict__ xr,      // res at col 2048
                            const float* __restrict__ A_log,
                            const float* __restrict__ D_param,
                            __bf16* __restrict__ ybf) {
  int tid = blockIdx.x * blockDim.x + threadIdx.x;   // 0 .. D_INNER*16-1
  int n = tid & 15;
  int d = tid >> 4;
  float a  = -__expf(A_log[(size_t)d * D_STATE + n]);
  float Dp = D_param[d];
  float s = 0.f;
  for (int t = 0; t < SEQ; ++t) {
    float dt = 0.f, u = 0.f;
    if (n == 0) {
      dt = delta[(size_t)t * D_INNER + d];
      u  = xc[(size_t)t * D_INNER + d];
    }
    dt = __shfl(dt, 0, 16);
    u  = __shfl(u, 0, 16);
    float Bn = xdbl[(size_t)t * XPROJ_N + DT_RANK + n];
    float Cn = xdbl[(size_t)t * XPROJ_N + DT_RANK + D_STATE + n];
    float dA = __expf(dt * a);
    s = dA * s + dt * Bn * u;
    float part = s * Cn;
    part += __shfl_xor(part, 1, 16);
    part += __shfl_xor(part, 2, 16);
    part += __shfl_xor(part, 4, 16);
    part += __shfl_xor(part, 8, 16);
    if (n == 0) {
      float r = xr[(size_t)t * (2 * D_INNER) + D_INNER + d];
      float g = r / (1.f + __expf(-r));              // silu(res)
      float yv = (part + u * Dp) * g;
      ybf[(size_t)t * D_INNER + d] = (__bf16)yv;
    }
  }
}

// ---------------------------------------------------------------------------
// Host-side orchestration
// ---------------------------------------------------------------------------
extern "C" void kernel_launch(void* const* d_in, const int* in_sizes, int n_in,
                              void* d_out, int out_size, void* d_ws, size_t ws_size,
                              hipStream_t stream) {
  (void)in_sizes; (void)n_in; (void)out_size; (void)ws_size;
  const float* x       = (const float*)d_in[0];   // [2048,1024]
  const float* norm_w  = (const float*)d_in[1];   // [1024]
  const float* in_W    = (const float*)d_in[2];   // [1024,4096]
  const float* conv_W  = (const float*)d_in[3];   // [4,1,2048]
  const float* conv_b  = (const float*)d_in[4];   // [2048]
  const float* xproj_W = (const float*)d_in[5];   // [2048,96]
  const float* dt_W    = (const float*)d_in[6];   // [64,2048]
  const float* dt_b    = (const float*)d_in[7];   // [2048]
  const float* A_log   = (const float*)d_in[8];   // [2048,16]
  const float* D_param = (const float*)d_in[9];   // [2048]
  const float* out_W   = (const float*)d_in[10];  // [2048,1024]
  float* out = (float*)d_out;                     // [2048,1024]

  // workspace carve-up (all sizes 256B-aligned). Bt buffers are transposed.
  char* w = (char*)d_ws;
  size_t off = 0;
  __bf16* ws_h     = (__bf16*)(w + off); off += (size_t)SEQ * D_MODEL * 2;        // 4 MB
  __bf16* ws_inWt  = (__bf16*)(w + off); off += (size_t)(2 * D_INNER) * D_MODEL * 2; // 8 MB [4096,1024]
  float*  ws_xr    = (float*) (w + off); off += (size_t)SEQ * 2 * D_INNER * 4;    // 32 MB
  float*  ws_xc    = (float*) (w + off); off += (size_t)SEQ * D_INNER * 4;        // 16 MB
  __bf16* ws_xcbf  = (__bf16*)(w + off); off += (size_t)SEQ * D_INNER * 2;        // 8 MB
  __bf16* ws_xpWt  = (__bf16*)(w + off); off += (size_t)XPROJ_N * D_INNER * 2;    // [96,2048]
  float*  ws_xdbl  = (float*) (w + off); off += (size_t)SEQ * XPROJ_N * 4;
  __bf16* ws_dlt   = (__bf16*)(w + off); off += (size_t)SEQ * DT_RANK * 2;
  __bf16* ws_dtWt  = (__bf16*)(w + off); off += (size_t)D_INNER * DT_RANK * 2;    // [2048,64]
  float*  ws_delta = (float*) (w + off); off += (size_t)SEQ * D_INNER * 4;        // 16 MB
  __bf16* ws_ybf   = (__bf16*)(w + off); off += (size_t)SEQ * D_INNER * 2;        // 8 MB
  __bf16* ws_outWt = (__bf16*)(w + off); off += (size_t)D_MODEL * D_INNER * 2;    // [1024,2048]

  // 1) weights -> transposed bf16 (N-major) for b128 B-fragment loads
  transpose_to_bf16_kernel<<<dim3((2 * D_INNER) / 32, D_MODEL / 32), 256, 0, stream>>>(
      in_W, ws_inWt, D_MODEL, 2 * D_INNER);
  transpose_to_bf16_kernel<<<dim3(XPROJ_N / 32, D_INNER / 32), 256, 0, stream>>>(
      xproj_W, ws_xpWt, D_INNER, XPROJ_N);
  transpose_to_bf16_kernel<<<dim3(D_INNER / 32, DT_RANK / 32), 256, 0, stream>>>(
      dt_W, ws_dtWt, DT_RANK, D_INNER);
  transpose_to_bf16_kernel<<<dim3(D_MODEL / 32, D_INNER / 32), 256, 0, stream>>>(
      out_W, ws_outWt, D_INNER, D_MODEL);

  // 2) RMSNorm -> h (bf16)
  rmsnorm_kernel<<<SEQ, 256, 0, stream>>>(x, norm_w, ws_h);

  // 3) in_proj: xr[2048,4096] = h @ in_W
  wmma_gemm_bf16_kernel<0><<<dim3((2 * D_INNER) / 128, SEQ / 16), 128, 0, stream>>>(
      ws_h, ws_inWt, ws_xr, SEQ, 2 * D_INNER, D_MODEL, nullptr, nullptr);

  // 4) depthwise causal conv + SiLU -> xc (f32 + bf16)
  conv_silu_kernel<<<(SEQ * D_INNER) / 256, 256, 0, stream>>>(
      ws_xr, conv_W, conv_b, ws_xc, ws_xcbf);

  // 5) x_proj: xdbl[2048,96] = xc @ xproj_W
  wmma_gemm_bf16_kernel<0><<<dim3(1, SEQ / 16), 128, 0, stream>>>(
      ws_xcbf, ws_xpWt, ws_xdbl, SEQ, XPROJ_N, D_INNER, nullptr, nullptr);

  // 6) dlt slice -> bf16
  slice_f32_to_bf16_kernel<<<(SEQ * DT_RANK) / 256, 256, 0, stream>>>(
      ws_xdbl, ws_dlt, SEQ, XPROJ_N, DT_RANK);

  // 7) delta = softplus(dlt @ dt_W + dt_b)
  wmma_gemm_bf16_kernel<1><<<dim3(D_INNER / 128, SEQ / 16), 128, 0, stream>>>(
      ws_dlt, ws_dtWt, ws_delta, SEQ, D_INNER, DT_RANK, dt_b, nullptr);

  // 8) selective scan + D skip + gate -> y (bf16)
  scan_kernel<<<(D_INNER * D_STATE) / 256, 256, 0, stream>>>(
      ws_delta, ws_xc, ws_xdbl, ws_xr, A_log, D_param, ws_ybf);

  // 9) out_proj + residual: out = y @ out_W + x
  wmma_gemm_bf16_kernel<2><<<dim3(D_MODEL / 128, SEQ / 16), 128, 0, stream>>>(
      ws_ybf, ws_outWt, out, SEQ, D_MODEL, D_INNER, nullptr, x);
}
